// SGHFM_69217692942833
// MI455X (gfx1250) — matrix operand order repaired and checked by reading, not compiled, not run on hardware
//
#include <hip/hip_runtime.h>
#include <hip/hip_bf16.h>

typedef unsigned short ushort_t;
typedef __attribute__((ext_vector_type(16))) __bf16 v16bf;
typedef __attribute__((ext_vector_type(8)))  __bf16 v8bf;
typedef __attribute__((ext_vector_type(8)))  float  v8f;

#define B_CONST 2
#define T_CONST 2048
#define DENSE   1024
#define SPARSE  1024
#define NHEADS  8
#define HDIM    128
#define WIN     64
#define KSEL    32
#define RNK     32

__device__ __forceinline__ unsigned short f32_to_bf16_rne(float f) {
    unsigned int u = __float_as_uint(f);
    u += 0x7FFFu + ((u >> 16) & 1u);
    return (unsigned short)(u >> 16);
}

// ---------------- conversion / transpose kernels ----------------

__global__ void f2bf_kernel(const float* __restrict__ src, ushort_t* __restrict__ dst, int n) {
    int i = blockIdx.x * blockDim.x + threadIdx.x;
    if (i < n) dst[i] = f32_to_bf16_rne(src[i]);
}

// enc (NH, DENSE, HD) -> dst[col*DENSE + d], col = n*HD + e
__global__ void encT_kernel(const float* __restrict__ src, ushort_t* __restrict__ dst) {
    int idx = blockIdx.x * blockDim.x + threadIdx.x;       // 0 .. 1024*1024-1
    int col = idx >> 10;
    int d   = idx & 1023;
    int n = col >> 7, e = col & 127;
    dst[idx] = f32_to_bf16_rne(src[((size_t)n * DENSE + d) * HDIM + e]);
}

// decoder (K=1024, N=1024) -> dst[col*K + k]
__global__ void decT_kernel(const float* __restrict__ src, ushort_t* __restrict__ dst) {
    int idx = blockIdx.x * blockDim.x + threadIdx.x;
    int col = idx >> 10;
    int k   = idx & 1023;
    dst[idx] = f32_to_bf16_rne(src[(size_t)k * DENSE + col]);
}

// ---------------- WMMA bf16 GEMM: C(M,N) = A(M,K) * Bt(N,K)^T ----------------
// One wave computes a 64x64 tile: 4x4 grid of 16x16 WMMA accumulators.
// Per K-step (32): 4 A frags + 4 B frags (16 b128 loads) -> 16 v_wmma.

__device__ __forceinline__ v16bf frag_from(const ushort_t* p0, const ushort_t* p1) {
    v8bf lo = *(const v8bf*)p0;
    v8bf hi = *(const v8bf*)p1;
    v16bf r;
#pragma unroll
    for (int i = 0; i < 8; ++i) { r[i] = lo[i]; r[i + 8] = hi[i]; }
    return r;
}

__global__ void __launch_bounds__(32)
gemm_bf16_wmma(const ushort_t* __restrict__ A, const ushort_t* __restrict__ Bt,
               float* __restrict__ C, int M, int N, int Kd, int do_relu)
{
    const int lane = threadIdx.x & 31;
    const int half = lane >> 4;
    const int r    = lane & 15;
    const int mt = blockIdx.y * 64;
    const int nt = blockIdx.x * 64;

    const ushort_t* ap[4];
    const ushort_t* bp[4];
#pragma unroll
    for (int i = 0; i < 4; ++i) {
        ap[i] = A  + (size_t)(mt + 16 * i + r) * Kd;
        bp[i] = Bt + (size_t)(nt + 16 * i + r) * Kd;
    }

    v8f acc[4][4];
#pragma unroll
    for (int i = 0; i < 4; ++i)
#pragma unroll
        for (int j = 0; j < 4; ++j)
            acc[i][j] = (v8f){0,0,0,0,0,0,0,0};

    for (int k0 = 0; k0 < Kd; k0 += 32) {
        const int ka = k0 + 8 * half;   // A: j<8 -> K=8h+j ; j>=8 -> K=16+8h+(j-8)
        const int kb = k0 + 8 * half;   // B: j<8 -> K=8h+j ; j>=8 -> K=16+8h+(j-8)
        v16bf af[4], bf[4];
#pragma unroll
        for (int i = 0; i < 4; ++i) af[i] = frag_from(ap[i] + ka, ap[i] + ka + 16);
#pragma unroll
        for (int j = 0; j < 4; ++j) bf[j] = frag_from(bp[j] + kb, bp[j] + kb + 16);
#pragma unroll
        for (int i = 0; i < 4; ++i)
#pragma unroll
            for (int j = 0; j < 4; ++j)
                acc[i][j] = __builtin_amdgcn_wmma_f32_16x16x32_bf16(
                    false, af[i], false, bf[j], (short)0, acc[i][j], false, false);
    }

#pragma unroll
    for (int i = 0; i < 4; ++i) {
        const int row0 = mt + 16 * i + 8 * half;   // C/D layout: half selects M+0/M+8
#pragma unroll
        for (int j = 0; j < 4; ++j) {
            const int col0 = nt + 16 * j + r;
#pragma unroll
            for (int e = 0; e < 8; ++e) {
                float v = acc[i][j][e];
                if (do_relu) v = fmaxf(v, 0.f);
                C[(size_t)(row0 + e) * N + col0] = v;
            }
        }
    }
}

// ---------------- signature reduction: sig[b][d] = meanT + 0.5*mean(::8) ----------------

__global__ void sig_kernel(const float* __restrict__ qs, float* __restrict__ sig) {
    int idx = blockIdx.x * blockDim.x + threadIdx.x;   // 0 .. B*SPARSE-1
    int d = idx & (SPARSE - 1);
    int b = idx >> 10;
    float s = 0.f, s8 = 0.f;
    for (int t = 0; t < T_CONST; ++t) {
        float v = qs[((size_t)(b * T_CONST + t)) * SPARSE + d];
        s += v;
        if ((t & 7) == 0) s8 += v;
    }
    sig[idx] = s * (1.0f / T_CONST) + 0.5f * s8 * (8.0f / T_CONST);
}

// ---------------- metric MLP (exact GELU, softplus) ----------------

__global__ void __launch_bounds__(512)
metric_mlp(const float* __restrict__ sig, const float* __restrict__ w1, const float* __restrict__ b1,
           const float* __restrict__ w2, const float* __restrict__ b2, const float* __restrict__ base,
           float* __restrict__ metric)
{
    const int b = blockIdx.x;
    const int j = threadIdx.x;            // 0..511
    __shared__ float hsh[SPARSE / 2];

    float acc = b1[j];
    for (int d = 0; d < SPARSE; ++d)
        acc += sig[b * SPARSE + d] * w1[(size_t)d * (SPARSE / 2) + j];
    hsh[j] = 0.5f * acc * (1.0f + erff(acc * 0.70710678118654752f)); // exact gelu
    __syncthreads();

    for (int c = j; c < SPARSE; c += 512) {
        float m = b2[c];
        for (int k = 0; k < SPARSE / 2; ++k)
            m += hsh[k] * w2[(size_t)k * SPARSE + c];
        float v = base[c] + 0.1f * m;
        // stable softplus
        float sp = fmaxf(v, 0.f) + log1pf(expf(-fabsf(v)));
        metric[b * SPARSE + c] = sp;
    }
}

// ---------------- hebbian matrix precompute: hebW[h][e][d] = U[h] @ V[h] ----------------

__global__ void hebw_kernel(const float* __restrict__ U, const float* __restrict__ V,
                            float* __restrict__ hebW)
{
    int idx = blockIdx.x * blockDim.x + threadIdx.x;   // 0 .. NH*HD*HD-1
    int h = idx >> 14;
    int rem = idx & 16383;
    int e = rem >> 7;
    int d = rem & 127;
    float s = 0.f;
    for (int r = 0; r < RNK; ++r)
        s += U[((size_t)h * HDIM + e) * RNK + r] * V[((size_t)h * RNK + r) * HDIM + d];
    hebW[idx] = s;
}

// ---------------- windowed metric attention: one wave per (b,t,h) ----------------

__global__ void __launch_bounds__(256)
attention_kernel(const float* __restrict__ qs, const float* __restrict__ vs,
                 const float* __restrict__ metric, const float* __restrict__ hebW,
                 ushort_t* __restrict__ osb)
{
    const int lane = threadIdx.x & 31;
    const int gw = blockIdx.x * 8 + (threadIdx.x >> 5);
    const int h = gw & (NHEADS - 1);
    const int t = (gw >> 3) & (T_CONST - 1);
    const int b = gw >> 14;                      // NH*T = 16384

    const size_t tokbase = ((size_t)(b * T_CONST + t)) * SPARSE + h * HDIM;
    const float4 q4 = *(const float4*)(qs + tokbase + lane * 4);
    const float4 m4 = *(const float4*)(metric + (size_t)b * SPARSE + h * HDIM + lane * 4);
    float q[4] = {q4.x, q4.y, q4.z, q4.w};
    float m[4] = {m4.x, m4.y, m4.z, m4.w};

    const float INF = __builtin_inff();
    float d0 = INF, d1 = INF;

    // cooperative distances over the causal window
    for (int w = 0; w < WIN; ++w) {
        int tk  = t - w;
        int tkc = tk < 0 ? 0 : tk;
        const float4 k4 = *(const float4*)(qs + ((size_t)(b * T_CONST + tkc)) * SPARSE + h * HDIM + lane * 4);
        float kk[4] = {k4.x, k4.y, k4.z, k4.w};
        float s = 0.f;
#pragma unroll
        for (int i = 0; i < 4; ++i) { float df = kk[i] - q[i]; s += m[i] * df * df; }
#pragma unroll
        for (int off = 16; off > 0; off >>= 1) s += __shfl_xor(s, off, 32);
        float dd = (tk >= 0) ? sqrtf(s + 1e-8f) : INF;
        if (w < 32) { if (lane == w)      d0 = dd; }
        else        { if (lane == w - 32) d1 = dd; }
    }

    // exact top-K=32 of 64 via rank counting (tie-break by lower window index)
    int r0 = 0, r1 = 0;
    for (int s = 0; s < 32; ++s) {
        float c0 = __shfl(d0, s, 32);   // candidate index s
        float c1 = __shfl(d1, s, 32);   // candidate index s+32
        r0 += ((c0 < d0) || (c0 == d0 && s < lane)) ? 1 : 0;
        // candidate s+32 vs my index lane: s+32 > lane always, ties lose -> only strict
        r0 += (c1 < d0) ? 1 : 0;
        // candidate s vs my index lane+32: s < lane+32 always, ties win
        r1 += ((c0 < d1) || (c0 == d1)) ? 1 : 0;
        r1 += ((c1 < d1) || (c1 == d1 && s < lane)) ? 1 : 0;
    }
    float e0 = (r0 < KSEL) ? expf(-d0) : 0.f;
    float e1 = (r1 < KSEL) ? expf(-d1) : 0.f;

    float es = e0 + e1;
#pragma unroll
    for (int off = 16; off > 0; off >>= 1) es += __shfl_xor(es, off, 32);

    float ctx[4] = {0.f, 0.f, 0.f, 0.f};
    for (int w = 0; w < 32; ++w) {
        float ew = __shfl(e0, w, 32);
        if (ew > 0.f) {  // ew>0 implies t-w >= 0
            const float4 vv = *(const float4*)(vs + ((size_t)(b * T_CONST + (t - w))) * SPARSE + h * HDIM + lane * 4);
            ctx[0] += ew * vv.x; ctx[1] += ew * vv.y; ctx[2] += ew * vv.z; ctx[3] += ew * vv.w;
        }
    }
    for (int w = 32; w < 64; ++w) {
        float ew = __shfl(e1, w - 32, 32);
        if (ew > 0.f) {
            const float4 vv = *(const float4*)(vs + ((size_t)(b * T_CONST + (t - w))) * SPARSE + h * HDIM + lane * 4);
            ctx[0] += ew * vv.x; ctx[1] += ew * vv.y; ctx[2] += ew * vv.z; ctx[3] += ew * vv.w;
        }
    }
    float inv = 1.f / (es + 1e-8f);
#pragma unroll
    for (int i = 0; i < 4; ++i) ctx[i] *= inv;

    // gated hebbian term (gate wave-uniform after reduction)
    float qq = q[0]*q[0] + q[1]*q[1] + q[2]*q[2] + q[3]*q[3];
#pragma unroll
    for (int off = 16; off > 0; off >>= 1) qq += __shfl_xor(qq, off, 32);
    float qn = sqrtf(qq);
    if (qn > 0.2f) {
        float invn = 1.f / fmaxf(qn, 1e-12f);
        float qu[4] = {q[0]*invn, q[1]*invn, q[2]*invn, q[3]*invn};
        float hc[4] = {0.f, 0.f, 0.f, 0.f};
        const float* hb = hebW + (size_t)h * HDIM * HDIM;
        for (int s = 0; s < 32; ++s) {
            float a0 = __shfl(qu[0], s, 32);
            float a1 = __shfl(qu[1], s, 32);
            float a2 = __shfl(qu[2], s, 32);
            float a3 = __shfl(qu[3], s, 32);
            const float* rowp = hb + (size_t)(s * 4) * HDIM + lane * 4;
            const float4 w0 = *(const float4*)(rowp);
            const float4 w1 = *(const float4*)(rowp + HDIM);
            const float4 w2 = *(const float4*)(rowp + 2 * HDIM);
            const float4 w3 = *(const float4*)(rowp + 3 * HDIM);
            hc[0] += a0*w0.x + a1*w1.x + a2*w2.x + a3*w3.x;
            hc[1] += a0*w0.y + a1*w1.y + a2*w2.y + a3*w3.y;
            hc[2] += a0*w0.z + a1*w1.z + a2*w2.z + a3*w3.z;
            hc[3] += a0*w0.w + a1*w1.w + a2*w2.w + a3*w3.w;
        }
#pragma unroll
        for (int i = 0; i < 4; ++i) ctx[i] += 0.1f * hc[i];
    }

#pragma unroll
    for (int i = 0; i < 4; ++i)
        osb[tokbase + lane * 4 + i] = f32_to_bf16_rne(q[i] * ctx[i]);
}

// ---------------- launcher ----------------

extern "C" void kernel_launch(void* const* d_in, const int* in_sizes, int n_in,
                              void* d_out, int out_size, void* d_ws, size_t ws_size,
                              hipStream_t stream)
{
    const float* x       = (const float*)d_in[0];
    const float* enc_q   = (const float*)d_in[1];
    const float* enc_v   = (const float*)d_in[2];
    const float* w1      = (const float*)d_in[3];
    const float* b1      = (const float*)d_in[4];
    const float* w2      = (const float*)d_in[5];
    const float* b2      = (const float*)d_in[6];
    const float* base_m  = (const float*)d_in[7];
    const float* hebb_U  = (const float*)d_in[8];
    const float* hebb_V  = (const float*)d_in[9];
    const float* decoder = (const float*)d_in[10];
    float* out = (float*)d_out;
    (void)in_sizes; (void)n_in; (void)out_size; (void)ws_size;

    const int BT = B_CONST * T_CONST;   // 4096
    char* wsp = (char*)d_ws;
    size_t off = 0;
    auto alloc = [&](size_t bytes) -> void* {
        void* p = wsp + off;
        off += (bytes + 255) & ~(size_t)255;
        return p;
    };
    ushort_t* xb   = (ushort_t*)alloc((size_t)BT * DENSE * 2);
    ushort_t* wqt  = (ushort_t*)alloc((size_t)SPARSE * DENSE * 2);
    ushort_t* wvt  = (ushort_t*)alloc((size_t)SPARSE * DENSE * 2);
    ushort_t* dect = (ushort_t*)alloc((size_t)DENSE * SPARSE * 2);
    float*    qsb  = (float*)alloc((size_t)BT * SPARSE * 4);
    float*    vsb  = (float*)alloc((size_t)BT * SPARSE * 4);
    ushort_t* osb  = (ushort_t*)alloc((size_t)BT * SPARSE * 2);
    float*    sig  = (float*)alloc((size_t)B_CONST * SPARSE * 4);
    float*    met  = (float*)alloc((size_t)B_CONST * SPARSE * 4);
    float*    hbw  = (float*)alloc((size_t)NHEADS * HDIM * HDIM * 4);

    // 1) precision conversion + weight transposes (bf16 [col][k] layouts)
    f2bf_kernel<<<(BT * DENSE) / 256, 256, 0, stream>>>(x, xb, BT * DENSE);
    encT_kernel<<<(SPARSE * DENSE) / 256, 256, 0, stream>>>(enc_q, wqt);
    encT_kernel<<<(SPARSE * DENSE) / 256, 256, 0, stream>>>(enc_v, wvt);
    decT_kernel<<<(DENSE * SPARSE) / 256, 256, 0, stream>>>(decoder, dect);

    // 2) encoder GEMMs with fused ReLU (bf16 WMMA, fp32 accum), 64x64 tile per wave
    dim3 ggrid(SPARSE / 64, BT / 64);
    gemm_bf16_wmma<<<ggrid, 32, 0, stream>>>(xb, wqt, qsb, BT, SPARSE, DENSE, 1);
    gemm_bf16_wmma<<<ggrid, 32, 0, stream>>>(xb, wvt, vsb, BT, SPARSE, DENSE, 1);

    // 3) metric path
    sig_kernel<<<(B_CONST * SPARSE) / 256, 256, 0, stream>>>(qsb, sig);
    metric_mlp<<<B_CONST, 512, 0, stream>>>(sig, w1, b1, w2, b2, base_m, met);

    // 4) hebbian matrices
    hebw_kernel<<<(NHEADS * HDIM * HDIM) / 256, 256, 0, stream>>>(hebb_U, hebb_V, hbw);

    // 5) windowed attention + gating + hebbian; writes bf16 out_sparse
    attention_kernel<<<(B_CONST * T_CONST * NHEADS) / 8, 256, 0, stream>>>(qsb, vsb, met, hbw, osb);

    // 6) decoder GEMM -> fp32 output
    gemm_bf16_wmma<<<dim3(DENSE / 64, BT / 64), 32, 0, stream>>>(osb, dect, out, BT, DENSE, SPARSE, 0);
}
